// ScaledDotProductAttention_60301340836528
// MI455X (gfx1250) — compile-verified, matrix operand order checked by reading, not run on hardware
//
#include <hip/hip_runtime.h>

typedef __attribute__((ext_vector_type(16))) __bf16          v16bf;
typedef __attribute__((ext_vector_type(8)))  float           v8f;
typedef __attribute__((ext_vector_type(8)))  unsigned short  u16x8;

constexpr int kB      = 16;
constexpr int kS      = 2048;
constexpr int kD      = 128;
constexpr int kWaves  = 8;            // 256 threads / wave32
constexpr int kQTile  = 16;           // q rows per wave
constexpr int kQBlock = kWaves * kQTile;   // 128 q rows per workgroup
constexpr int kKBlock = 32;           // keys per iteration
constexpr int kNB     = kS / kKBlock; // 64 key blocks
constexpr int kPitchK = 136;          // sK row pitch in shorts (272B: 16B-aligned, bank-spread)
constexpr int kPitchV = 40;           // sVT row pitch in shorts (80B)
constexpr int kPitchP = 40;           // sP row pitch in shorts (80B)

__device__ __forceinline__ unsigned short f2bf(float x) {
  unsigned int u = __builtin_bit_cast(unsigned int, x);
  u += 0x7fffu + ((u >> 16) & 1u);            // round-to-nearest-even
  return (unsigned short)(u >> 16);
}

__device__ __forceinline__ v8f zero8() {
  v8f z;
#pragma unroll
  for (int i = 0; i < 8; ++i) z[i] = 0.0f;
  return z;
}

// Build a 16-element bf16 frag from two 16-byte LDS runs (ds_load_b128 x2).
__device__ __forceinline__ v16bf frag_lds(const unsigned short* p0,
                                          const unsigned short* p1) {
  union { u16x8 h[2]; v16bf v; } r;
  r.h[0] = *(const u16x8*)p0;
  r.h[1] = *(const u16x8*)p1;
  return r.v;
}

// Build a frag from two 8-float global runs, scaled, converted to bf16.
__device__ __forceinline__ v16bf frag_global_scaled(const float* p0,
                                                    const float* p1, float s) {
  union { unsigned short u[16]; v16bf v; } r;
  float4 a = *(const float4*)(p0);
  float4 b = *(const float4*)(p0 + 4);
  float4 c = *(const float4*)(p1);
  float4 d = *(const float4*)(p1 + 4);
  r.u[0]  = f2bf(a.x * s); r.u[1]  = f2bf(a.y * s);
  r.u[2]  = f2bf(a.z * s); r.u[3]  = f2bf(a.w * s);
  r.u[4]  = f2bf(b.x * s); r.u[5]  = f2bf(b.y * s);
  r.u[6]  = f2bf(b.z * s); r.u[7]  = f2bf(b.w * s);
  r.u[8]  = f2bf(c.x * s); r.u[9]  = f2bf(c.y * s);
  r.u[10] = f2bf(c.z * s); r.u[11] = f2bf(c.w * s);
  r.u[12] = f2bf(d.x * s); r.u[13] = f2bf(d.y * s);
  r.u[14] = f2bf(d.z * s); r.u[15] = f2bf(d.w * s);
  return r.v;
}

__global__ __launch_bounds__(256, 1)
void fa_fwd_bf16_wmma(const float* __restrict__ Q, const float* __restrict__ K,
                      const float* __restrict__ V, const unsigned char* __restrict__ M,
                      float* __restrict__ O) {
  __shared__ __align__(16) unsigned short sK [kKBlock * kPitchK];        // K block, [key][d]
  __shared__ __align__(16) unsigned short sVT[kD * kPitchV];             // V block transposed, [d][key]
  __shared__ __align__(16) unsigned short sP [kWaves][kQTile * kPitchP]; // per-wave P tile, [m][k]

  const int tid  = threadIdx.x;
  const int wave = tid >> 5;
  const int lane = tid & 31;
  const int half = lane >> 4;     // 0: lanes 0-15, 1: lanes 16-31
  const int n16  = lane & 15;

  const int b  = blockIdx.x >> 4;       // 16 q-blocks per batch
  const int qb = blockIdx.x & 15;
  const int q0 = qb * kQBlock + wave * kQTile;

  const float* Qp = Q + ((size_t)b * kS + q0) * kD;
  const float* Kp = K + (size_t)b * kS * kD;
  const float* Vp = V + (size_t)b * kS * kD;
  const unsigned char* Mp = M + (size_t)b * kS * kS;
  float* Op = O + ((size_t)b * kS + q0) * kD;

  // Base-2 softmax domain: fold log2(e) into the 1/sqrt(D) scale so every
  // exponential is a bare v_exp_f32 (exp2f) with no ln2 multiply.
  const float scale = 0.12751791097443183f;  // log2(e) / sqrt(128)

  // ---- Q tile as 4 bf16 A-frags (pre-scaled). A-layout 16-bit 16x32:
  // lane m = lane%16, element runs contiguous at K = half*8.. and 16+half*8..
  v16bf aq[4];
#pragma unroll
  for (int c = 0; c < 4; ++c) {
    const float* p0 = Qp + (size_t)n16 * kD + c * 32 + half * 8;
    aq[c] = frag_global_scaled(p0, p0 + 16, scale);
  }

  v8f oacc[8];
#pragma unroll
  for (int t = 0; t < 8; ++t) oacc[t] = zero8();
  float mrow[8], lrow[8];
#pragma unroll
  for (int r = 0; r < 8; ++r) { mrow[r] = -1e30f; lrow[r] = 0.0f; }

  // ---- Software-pipelined K/V staging: 256 threads x 16 floats cover 32x128.
  const int srow = tid >> 3;                 // 0..31 key row
  const int scol = (tid & 7) * 16;           // 0..112 d column base
  const float* kstream = Kp + (size_t)srow * kD + scol;
  const float* vstream = Vp + (size_t)srow * kD + scol;

  float4 kreg[4], vreg[4];
#pragma unroll
  for (int i = 0; i < 4; ++i) {              // prefetch block 0
    kreg[i] = *(const float4*)(kstream + i * 4);
    vreg[i] = *(const float4*)(vstream + i * 4);
  }

  for (int kb = 0; kb < kNB; ++kb) {
    __syncthreads();  // previous-iteration LDS reads done before restaging
    // Store the prefetched block (registers -> LDS, with f32 -> bf16 RNE).
#pragma unroll
    for (int i = 0; i < 4; ++i) {
      const int c = scol + i * 4;
      sK[srow * kPitchK + c + 0] = f2bf(kreg[i].x);
      sK[srow * kPitchK + c + 1] = f2bf(kreg[i].y);
      sK[srow * kPitchK + c + 2] = f2bf(kreg[i].z);
      sK[srow * kPitchK + c + 3] = f2bf(kreg[i].w);
      sVT[(c + 0) * kPitchV + srow] = f2bf(vreg[i].x);
      sVT[(c + 1) * kPitchV + srow] = f2bf(vreg[i].y);
      sVT[(c + 2) * kPitchV + srow] = f2bf(vreg[i].z);
      sVT[(c + 3) * kPitchV + srow] = f2bf(vreg[i].w);
    }
    __syncthreads();

    // Mask bytes for this lane's 16 score elements. Issued FIRST so the
    // softmax-phase s_wait_loadcnt for these bytes does not transitively
    // wait on the K/V prefetch (LOADcnt retires strictly in issue order).
    unsigned char mb0[8], mb1[8];
#pragma unroll
    for (int r = 0; r < 8; ++r) {
      const size_t mi = (size_t)(q0 + r + 8 * half) * kS
                      + (size_t)kb * kKBlock + n16;
      mb0[r] = Mp[mi];
      mb1[r] = Mp[mi + 16];
    }

    // Prefetch next key block AFTER the mask bytes: it only has to complete
    // by the next iteration's barrier, hiding under softmax + PV as well.
    if (kb + 1 < kNB) {
      const float* kn = kstream + (size_t)(kb + 1) * kKBlock * kD;
      const float* vn = vstream + (size_t)(kb + 1) * kKBlock * kD;
#pragma unroll
      for (int i = 0; i < 4; ++i) {
        kreg[i] = *(const float4*)(kn + i * 4);
        vreg[i] = *(const float4*)(vn + i * 4);
      }
    }

    // ---- S = Q @ K^T : two 16x16 tiles (keys n16 and 16+n16), K-dim = 128 = 4x32
    v8f c0 = zero8(), c1 = zero8();
#pragma unroll
    for (int c = 0; c < 4; ++c) {
      const unsigned short* b0p = &sK[n16 * kPitchK + c * 32 + half * 8];
      const unsigned short* b1p = &sK[(16 + n16) * kPitchK + c * 32 + half * 8];
      v16bf b0 = frag_lds(b0p, b0p + 16);
      v16bf b1 = frag_lds(b1p, b1p + 16);
      c0 = __builtin_amdgcn_wmma_f32_16x16x32_bf16(false, aq[c], false, b0,
                                                   (short)0, c0, false, false);
      c1 = __builtin_amdgcn_wmma_f32_16x16x32_bf16(false, aq[c], false, b1,
                                                   (short)0, c1, false, false);
    }

    // ---- mask + online softmax (base-2). C-layout: VGPR r -> row r+8*half,
    //      lane -> col n16.
#pragma unroll
    for (int r = 0; r < 8; ++r) {
      float s0 = c0[r], s1 = c1[r];
      if (mb0[r]) s0 = -1e30f;
      if (mb1[r]) s1 = -1e30f;
      float bm = fmaxf(s0, s1);
      bm = fmaxf(bm, __shfl_xor(bm, 1, 32));
      bm = fmaxf(bm, __shfl_xor(bm, 2, 32));
      bm = fmaxf(bm, __shfl_xor(bm, 4, 32));
      bm = fmaxf(bm, __shfl_xor(bm, 8, 32));
      const float mn    = fmaxf(mrow[r], bm);
      const float alpha = exp2f(mrow[r] - mn);
      mrow[r] = mn;
      const float p0 = exp2f(s0 - mn);
      const float p1 = exp2f(s1 - mn);
      c0[r] = p0; c1[r] = p1;
      float rs = p0 + p1;
      rs += __shfl_xor(rs, 1, 32);
      rs += __shfl_xor(rs, 2, 32);
      rs += __shfl_xor(rs, 4, 32);
      rs += __shfl_xor(rs, 8, 32);
      lrow[r] = lrow[r] * alpha + rs;
#pragma unroll
      for (int t = 0; t < 8; ++t) oacc[t][r] *= alpha;
    }

    // ---- P: C-layout -> A-layout via per-wave LDS bounce (same-wave DS is in-order).
    unsigned short* pw = sP[wave];
#pragma unroll
    for (int r = 0; r < 8; ++r) {
      const int row = r + 8 * half;
      pw[row * kPitchP + n16]      = f2bf(c0[r]);
      pw[row * kPitchP + 16 + n16] = f2bf(c1[r]);
    }
    __asm__ volatile("" ::: "memory");  // keep compiler from hoisting the reads
    const unsigned short* ppr = &pw[n16 * kPitchP + half * 8];
    v16bf ap = frag_lds(ppr, ppr + 16);

    // ---- O += P @ V : K-dim 32, 8 column tiles of D=128
#pragma unroll
    for (int t = 0; t < 8; ++t) {
      const unsigned short* vbp = &sVT[(t * 16 + n16) * kPitchV + half * 8];
      v16bf bv = frag_lds(vbp, vbp + 16);
      oacc[t] = __builtin_amdgcn_wmma_f32_16x16x32_bf16(false, ap, false, bv,
                                                        (short)0, oacc[t], false, false);
    }
  }

  // ---- normalize + store
#pragma unroll
  for (int r = 0; r < 8; ++r) {
    const float inv = 1.0f / lrow[r];
    const int row = r + 8 * half;
#pragma unroll
    for (int t = 0; t < 8; ++t) {
      Op[(size_t)row * kD + t * 16 + n16] = oacc[t][r] * inv;
    }
  }
}

extern "C" void kernel_launch(void* const* d_in, const int* in_sizes, int n_in,
                              void* d_out, int out_size, void* d_ws, size_t ws_size,
                              hipStream_t stream) {
  const float*         Q = (const float*)d_in[0];
  const float*         K = (const float*)d_in[1];
  const float*         V = (const float*)d_in[2];
  const unsigned char* M = (const unsigned char*)d_in[3];
  float*               O = (float*)d_out;

  const dim3 grid(kB * (kS / kQBlock));  // 16 batches x 16 q-blocks = 256 WGs
  const dim3 block(kWaves * 32);         // 8 waves
  fa_fwd_bf16_wmma<<<grid, block, 0, stream>>>(Q, K, V, M, O);

  (void)in_sizes; (void)n_in; (void)out_size; (void)d_ws; (void)ws_size;
}
